// MoE_47742856462430
// MI455X (gfx1250) — compile-verified, matrix operand order checked by reading, not run on hardware
//
#include <hip/hip_runtime.h>
#include <math.h>

// Problem constants (from reference setup_inputs)
#define TOK   8192   // B*L
#define DDIM  512
#define HDIM  1024
#define NEXP  16
#define TM    16     // token tile (WMMA M)

#define NFRAG1 (NEXP * (HDIM / 16) * (DDIM / 32))   // W1 fragments: 16384
#define NFRAG2 (NEXP * (DDIM / 16) * (HDIM / 32))   // W2 fragments: 16384

typedef __attribute__((ext_vector_type(16))) _Float16 v16h;
typedef __attribute__((ext_vector_type(8)))  float    v8f;

// K index of VGPR-pair v (0..7) of a 16-bit WMMA A/B fragment, given lane half.
__device__ __forceinline__ int kmap16(int v, int half) {
    return ((v < 4) ? 2 * v : 16 + 2 * (v - 4)) + 8 * half;
}

// Load one 16-half fragment (32 B) as two 128-bit vectors.
template <typename P>
__device__ __forceinline__ v16h load_frag(const P* p) {
    union { v16h v; float4 f[2]; } u;
    u.f[0] = ((const float4*)p)[0];
    u.f[1] = ((const float4*)p)[1];
    return u.v;
}

// ---------------------------------------------------------------------------
// Kernel 1: zero output accumulator + expert counters
// ---------------------------------------------------------------------------
__global__ __launch_bounds__(256)
void moe_zero_kernel(float* __restrict__ out, int n, int* __restrict__ counts) {
    int i = blockIdx.x * 256 + threadIdx.x;
    if (i < n)    out[i] = 0.0f;
    if (i < NEXP) counts[i] = 0;
}

// ---------------------------------------------------------------------------
// Kernel 2: repack W [E, Kd, N] f32 row-major -> f16 WMMA-B-fragment-major:
//   Wp[((e*(N/16)+nt)*(Kd/32)+kc)*512 + lane*16 + j]
// One wave per fragment; each lane writes its 32 contiguous bytes.
// ---------------------------------------------------------------------------
__global__ __launch_bounds__(256)
void moe_pack_kernel(const float* __restrict__ W, _Float16* __restrict__ Wp,
                     int N, int Kd, int nfrag) {
    int wave = threadIdx.x >> 5, lid = threadIdx.x & 31;
    int f = blockIdx.x * 8 + wave;
    if (f >= nfrag) return;
    int kc  = f % (Kd / 32);
    int rem = f / (Kd / 32);
    int nt  = rem % (N / 16);
    int e   = rem / (N / 16);
    int r = lid & 15, half = lid >> 4;

    _Float16* dst = Wp + (size_t)f * 512 + lid * 16;
    const float* src = W + (size_t)e * Kd * N + nt * 16 + r;
#pragma unroll
    for (int v = 0; v < 8; ++v) {
        int kb = kmap16(v, half) + kc * 32;
        dst[2 * v]     = (_Float16)src[(size_t)kb * N];
        dst[2 * v + 1] = (_Float16)src[(size_t)(kb + 1) * N];
    }
}

// ---------------------------------------------------------------------------
// Kernel 3: gate = softmax(x @ W_gate), top-2, append to per-expert lists
// ---------------------------------------------------------------------------
__global__ __launch_bounds__(256)
void moe_gate_kernel(const float* __restrict__ x, const float* __restrict__ Wg,
                     int* __restrict__ counts, int* __restrict__ toks,
                     float* __restrict__ gates) {
    __shared__ float wgl[DDIM * NEXP];           // 32 KB
    int tid = threadIdx.x;
    for (int i = tid; i < DDIM * NEXP; i += 256) wgl[i] = Wg[i];
    __syncthreads();

    int t = blockIdx.x * 256 + tid;              // one thread per token
    float acc[NEXP];
#pragma unroll
    for (int e = 0; e < NEXP; ++e) acc[e] = 0.0f;

    const float* xr = x + (size_t)t * DDIM;
    for (int d = 0; d < DDIM; ++d) {
        float xv = xr[d];
#pragma unroll
        for (int e = 0; e < NEXP; ++e) acc[e] += xv * wgl[d * NEXP + e];
    }
    float mx = acc[0];
#pragma unroll
    for (int e = 1; e < NEXP; ++e) mx = fmaxf(mx, acc[e]);
    float s = 0.0f;
#pragma unroll
    for (int e = 0; e < NEXP; ++e) { acc[e] = __expf(acc[e] - mx); s += acc[e]; }
    float inv = 1.0f / s;
    int i0 = 0; float p0 = acc[0];
#pragma unroll
    for (int e = 1; e < NEXP; ++e) if (acc[e] > p0) { p0 = acc[e]; i0 = e; }
    int i1 = (i0 == 0) ? 1 : 0; float p1 = acc[i1];
#pragma unroll
    for (int e = 0; e < NEXP; ++e)
        if (e != i0 && acc[e] > p1) { p1 = acc[e]; i1 = e; }
    p0 *= inv; p1 *= inv;

    int pos0 = atomicAdd(&counts[i0], 1);
    toks[i0 * TOK + pos0] = t;  gates[i0 * TOK + pos0] = p0;
    int pos1 = atomicAdd(&counts[i1], 1);
    toks[i1 * TOK + pos1] = t;  gates[i1 * TOK + pos1] = p1;
}

// ---------------------------------------------------------------------------
// Kernel 4: grouped expert FFN.  grid = (TOK/TM, NEXP), 256 threads (8 waves).
// A-fragments come from LDS in fragment order (2x ds_load_b128); B-fragments
// from pre-packed f16 weights (2x global_load_b128).
// ---------------------------------------------------------------------------
__global__ __launch_bounds__(256)
void moe_expert_kernel(const float* __restrict__ x,
                       const _Float16* __restrict__ W1p,
                       const _Float16* __restrict__ W2p,
                       const int* __restrict__ counts,
                       const int* __restrict__ toks,
                       const float* __restrict__ gates,
                       float* __restrict__ out) {
    const int e    = blockIdx.y;
    const int cnt  = counts[e];
    const int row0 = blockIdx.x * TM;
    if (row0 >= cnt) return;                      // block-uniform: EXEC stays full

    __shared__ _Float16 xsf[(DDIM / 32) * 512];   // 16 KB, x tile, fragment order
    __shared__ _Float16 hsf[(HDIM / 32) * 512];   // 32 KB, gelu(h) tile, fragment order
    __shared__ int   stok[TM];
    __shared__ float sg[TM];

    const int tid = threadIdx.x;
    if (tid < TM) {
        int rr = row0 + tid;
        if (rr < cnt) { stok[tid] = toks[e * TOK + rr]; sg[tid] = gates[e * TOK + rr]; }
        else          { stok[tid] = toks[e * TOK];      sg[tid] = 0.0f; } // pad: gate 0
    }
    __syncthreads();

    // Gather 16 token rows of x (coalesced along D) -> f16 fragment order in LDS
    for (int i = tid; i < TM * DDIM; i += 256) {
        int m = i >> 9, d = i & (DDIM - 1);
        float xv = x[(size_t)stok[m] * DDIM + d];
        int kc = d >> 5, kd = d & 31;
        int half = (kd >> 3) & 1;
        int j = ((kd & 16) >> 1) | (kd & 7);
        xsf[kc * 512 + (m + 16 * half) * 16 + j] = (_Float16)xv;
    }
    __syncthreads();

    const int wave = tid >> 5, lid = tid & 31;
    const int r = lid & 15, halfc = lid >> 4;
    const _Float16* W1e = W1p + (size_t)e * (HDIM / 16) * (DDIM / 32) * 512;
    const _Float16* W2e = W2p + (size_t)e * (DDIM / 16) * (HDIM / 32) * 512;

    // ---- GEMM1: h[16, H] = x_tile @ W1e, exact GELU, store to LDS frag order ----
    for (int nt = wave; nt < HDIM / 16; nt += 8) {
        const int n0 = nt * 16;
        const _Float16* wf = W1e + (size_t)nt * (DDIM / 32) * 512 + lid * 16;
        v8f c = {};
        for (int kc = 0; kc < DDIM / 32; ++kc) {
            __builtin_prefetch(wf + 512, 0, 0);   // next k-chunk fragment
            v16h A = load_frag(&xsf[kc * 512 + lid * 16]);
            v16h B = load_frag(wf);
            wf += 512;
            c = __builtin_amdgcn_wmma_f32_16x16x32_f16(false, A, false, B,
                                                       (short)0, c, false, false);
        }
        const int hcol = n0 + r;
        const int kc2 = hcol >> 5, kd = hcol & 31;
        const int hhalf = (kd >> 3) & 1;
        const int j = ((kd & 16) >> 1) | (kd & 7);
#pragma unroll
        for (int i = 0; i < 8; ++i) {
            int m = i + 8 * halfc;
            float hv = c[i];
            float g  = 0.5f * hv * (1.0f + erff(hv * 0.70710678f));  // exact gelu
            hsf[kc2 * 512 + (m + 16 * hhalf) * 16 + j] = (_Float16)g;
        }
    }
    __syncthreads();

    // ---- GEMM2: y[16, D] = h @ W2e, scale by gate, atomic-add into out ----
    for (int nt = wave; nt < DDIM / 16; nt += 8) {
        const int n0 = nt * 16;
        const _Float16* wf = W2e + (size_t)nt * (HDIM / 32) * 512 + lid * 16;
        v8f c = {};
        for (int kc = 0; kc < HDIM / 32; ++kc) {
            __builtin_prefetch(wf + 512, 0, 0);
            v16h A = load_frag(&hsf[kc * 512 + lid * 16]);
            v16h B = load_frag(wf);
            wf += 512;
            c = __builtin_amdgcn_wmma_f32_16x16x32_f16(false, A, false, B,
                                                       (short)0, c, false, false);
        }
#pragma unroll
        for (int i = 0; i < 8; ++i) {
            int m = i + 8 * halfc;
            atomicAdd(&out[(size_t)stok[m] * DDIM + n0 + r], sg[m] * c[i]);
        }
    }
}

// ---------------------------------------------------------------------------
extern "C" void kernel_launch(void* const* d_in, const int* in_sizes, int n_in,
                              void* d_out, int out_size, void* d_ws, size_t ws_size,
                              hipStream_t stream) {
    const float* x  = (const float*)d_in[0];   // [T, D]
    const float* Wg = (const float*)d_in[1];   // [D, E]
    const float* W1 = (const float*)d_in[2];   // [E, D, H]
    const float* W2 = (const float*)d_in[3];   // [E, H, D]
    float* out = (float*)d_out;                // [T, D]

    char* ws = (char*)d_ws;
    int*   counts = (int*)ws;                                       // 16 ints
    int*   toks   = (int*)(ws + 256);                               // E*T ints
    float* gates  = (float*)(ws + 256 + (size_t)NEXP * TOK * 4);    // E*T floats
    _Float16* W1p = (_Float16*)(ws + 256 + 2ull * NEXP * TOK * 4);  // 16.8 MB
    _Float16* W2p = W1p + (size_t)NFRAG1 * 512;                     // 16.8 MB

    moe_zero_kernel<<<(TOK * DDIM + 255) / 256, 256, 0, stream>>>(out, TOK * DDIM, counts);
    moe_pack_kernel<<<(NFRAG1 + 7) / 8, 256, 0, stream>>>(W1, W1p, HDIM, DDIM, NFRAG1);
    moe_pack_kernel<<<(NFRAG2 + 7) / 8, 256, 0, stream>>>(W2, W2p, DDIM, HDIM, NFRAG2);
    moe_gate_kernel<<<TOK / 256, 256, 0, stream>>>(x, Wg, counts, toks, gates);
    moe_expert_kernel<<<dim3(TOK / TM, NEXP), 256, 0, stream>>>(x, W1p, W2p, counts,
                                                                toks, gates, out);
}
